// QuantConv_2379411882586
// MI455X (gfx1250) — matrix operand order repaired and checked by reading, not compiled, not run on hardware
//
#include <hip/hip_runtime.h>

typedef __attribute__((ext_vector_type(16))) _Float16 v16h;
typedef __attribute__((ext_vector_type(8)))  _Float16 v8h;
typedef __attribute__((ext_vector_type(8)))  float    v8f;

#define C1   256
#define C2   512
#define HDIM 64
#define WDIM 64
#define NB   16
#define KTOT (C1 * 9)      // 2304
#define NK   (KTOT / 32)   // 72 K-steps
#define BLK_M 128          // output channels per block
#define BLK_N 64           // spatial positions per block (one full W row)
#define LDA   40           // padded LDS row stride (f16) for A tile
#define LDB   40           // padded LDS row stride (f16) for B tile

#ifndef __has_builtin
#define __has_builtin(x) 0
#endif
#if __has_builtin(__builtin_amdgcn_global_load_async_to_lds_b128) && \
    __has_builtin(__builtin_amdgcn_s_wait_asynccnt)
#define HAVE_ASYNC_LDS 1
#else
#define HAVE_ASYNC_LDS 0
#endif

#if HAVE_ASYNC_LDS
typedef int v4i __attribute__((vector_size(16)));
typedef v4i __attribute__((address_space(1))) v4i_g;   // global
typedef v4i __attribute__((address_space(3))) v4i_l;   // LDS
#endif

// ---------------------------------------------------------------------------
// Prep: per-output-channel int8 fake-quant of weights -> f16, packed so that
// K index = tap*256 + ci. Also folds BN into per-channel scale/bias.
// ---------------------------------------------------------------------------
__global__ __launch_bounds__(256) void qconv_prep_kernel(
    const float* __restrict__ w, const float* __restrict__ gamma,
    const float* __restrict__ beta, const float* __restrict__ mean,
    const float* __restrict__ var, _Float16* __restrict__ wpack,
    float* __restrict__ bnscale, float* __restrict__ bnbias)
{
  __shared__ float red[256];
  const int co = blockIdx.x;
  const int t  = threadIdx.x;
  const float* wc = w + (size_t)co * KTOT;

  float amax = 0.0f;
  for (int i = t; i < KTOT; i += 256) amax = fmaxf(amax, fabsf(wc[i]));
  red[t] = amax;
  __syncthreads();
  for (int s = 128; s > 0; s >>= 1) {
    if (t < s) red[t] = fmaxf(red[t], red[t + s]);
    __syncthreads();
  }
  const float scale = fmaxf(red[0] * (1.0f / 127.0f), 1e-12f);
  const float inv_s = 1.0f / scale;

  for (int i = t; i < KTOT; i += 256) {
    const int ci = i / 9;
    const int tap = i - ci * 9;   // kh*3+kw
    float q = fminf(fmaxf(rintf(wc[i] * inv_s), -127.0f), 127.0f) * scale;
    wpack[(size_t)co * KTOT + tap * C1 + ci] = (_Float16)q;
  }
  if (t == 0) {
    float s = gamma[co] * rsqrtf(var[co] + 1e-3f);
    bnscale[co] = s;
    bnbias[co]  = beta[co] - mean[co] * s;
  }
}

// ---------------------------------------------------------------------------
// Implicit-GEMM conv, double-buffered + software-pipelined.
// Block = 128 co x 64 w-positions, 8 wave32s, each wave 2x2 WMMA tiles.
// ---------------------------------------------------------------------------
__global__ __launch_bounds__(256, 2) void qconv_main_kernel(
    const float* __restrict__ x, const _Float16* __restrict__ wpack,
    const float* __restrict__ bnscale, const float* __restrict__ bnbias,
    const float* __restrict__ act_scale, float* __restrict__ out)
{
  __shared__ _Float16 As[2][BLK_M * LDA];   // 2 x (128 x 32 f16, padded)
  __shared__ _Float16 Bt[2][BLK_N * LDB];   // 2 x (64 x 32 f16, padded) [pos][k]

  const int t     = threadIdx.x;
  const int lane  = t & 31;
  const int wv    = t >> 5;
  const int waveM = wv & 3;              // 4 waves along M
  const int waveN = wv >> 2;             // 2 waves along N
  const int col   = lane & 15;
  const int hsel  = lane >> 4;

  const int nh    = blockIdx.x;          // 0..1023  (n*64 + h)
  const int n     = nh >> 6;
  const int h     = nh & 63;
  const int coBlk = blockIdx.y * BLK_M;

  // A staging mapping: 512 chunks of 8 f16, 2 per thread
  const int ar0 = t >> 2,          as0 = t & 3;
  const int ar1 = (t + 256) >> 2,  as1 = (t + 256) & 3;
  // B staging mapping: lane-contiguous in w (coalesced global loads)
  const int bpos = t & 63;
  const int bks  = t >> 6;

  v8f acc00 = {}, acc01 = {}, acc10 = {}, acc11 = {};

  // ---- pipeline helpers (all straight-line, branch-free) ----
  const _Float16* wbase = wpack + (size_t)coBlk * KTOT;

#if HAVE_ASYNC_LDS
  // DMA weights straight into LDS via GLOBAL_LOAD_ASYNC_TO_LDS_B128 (ASYNCcnt)
  #define STAGE_A(buf, kk)                                                      \
    do {                                                                        \
      const int tap_ = (kk) >> 3, ciB_ = ((kk) & 7) << 5;                       \
      const _Float16* g_ = wbase + tap_ * C1 + ciB_;                            \
      __builtin_amdgcn_global_load_async_to_lds_b128(                           \
          (v4i_g*)(g_ + (size_t)ar0 * KTOT + as0 * 8),                          \
          (v4i_l*)(uintptr_t)&As[buf][ar0 * LDA + as0 * 8], 0, 0);              \
      __builtin_amdgcn_global_load_async_to_lds_b128(                           \
          (v4i_g*)(g_ + (size_t)ar1 * KTOT + as1 * 8),                          \
          (v4i_l*)(uintptr_t)&As[buf][ar1 * LDA + as1 * 8], 0, 0);              \
    } while (0)
  #define WAIT_ASYNC() __builtin_amdgcn_s_wait_asynccnt(0)
#else
  v8h a0r = {}, a1r = {};
  #define STAGE_A_LOAD(kk)                                                      \
    do {                                                                        \
      const int tap_ = (kk) >> 3, ciB_ = ((kk) & 7) << 5;                       \
      const _Float16* g_ = wbase + tap_ * C1 + ciB_;                            \
      a0r = *(const v8h*)(g_ + (size_t)ar0 * KTOT + as0 * 8);                   \
      a1r = *(const v8h*)(g_ + (size_t)ar1 * KTOT + as1 * 8);                   \
    } while (0)
  #define STAGE_A_STORE(buf)                                                    \
    do {                                                                        \
      *(v8h*)&As[buf][ar0 * LDA + as0 * 8] = a0r;                               \
      *(v8h*)&As[buf][ar1 * LDA + as1 * 8] = a1r;                               \
    } while (0)
  #define WAIT_ASYNC() ((void)0)
#endif

  float breg[8];
  bool  okB = false;
  // unconditional loads from clamped addresses; zeroing happens at LDS store
  #define STAGE_B_LOAD(kk)                                                      \
    do {                                                                        \
      const int tap_ = (kk) >> 3, ciB_ = ((kk) & 7) << 5;                       \
      const int dh_ = tap_ / 3 - 1, dw_ = tap_ - (tap_ / 3) * 3 - 1;            \
      const int hh_ = h + dh_, ww_ = bpos + dw_;                                \
      okB = (hh_ >= 0) && (hh_ < HDIM) && (ww_ >= 0) && (ww_ < WDIM);           \
      const int hcl_ = min(max(hh_, 0), HDIM - 1);                              \
      const int wcl_ = min(max(ww_, 0), WDIM - 1);                              \
      const float* g_ =                                                         \
          x + (((size_t)n * C1 + (ciB_ + bks * 8)) * HDIM + hcl_) * WDIM + wcl_;\
      _Pragma("unroll")                                                         \
      for (int i_ = 0; i_ < 8; ++i_) breg[i_] = g_[(size_t)i_ * (HDIM * WDIM)]; \
    } while (0)
  #define STAGE_B_STORE(buf)                                                    \
    do {                                                                        \
      v8h bv_;                                                                  \
      _Pragma("unroll")                                                         \
      for (int i_ = 0; i_ < 8; ++i_)                                            \
        bv_[i_] = (_Float16)(okB ? breg[i_] : 0.0f);                            \
      *(v8h*)&Bt[buf][bpos * LDB + bks * 8] = bv_;                              \
    } while (0)

  // ---- prologue: fill buffer 0, preload regs for step 1 ----
#if HAVE_ASYNC_LDS
  STAGE_A(0, 0);
#else
  STAGE_A_LOAD(0);
  STAGE_A_STORE(0);
  STAGE_A_LOAD(1);
#endif
  STAGE_B_LOAD(0);
  STAGE_B_STORE(0);
  STAGE_B_LOAD(1);
  WAIT_ASYNC();
  __syncthreads();

  for (int kk = 0; kk < NK; ++kk) {
    const int cur = kk & 1;
    const int nxt = cur ^ 1;

    // stage kk+1 into the alternate buffer (overlaps with this step's WMMAs)
    if (kk + 1 < NK) {
#if HAVE_ASYNC_LDS
      STAGE_A(nxt, kk + 1);
#else
      STAGE_A_STORE(nxt);
#endif
      STAGE_B_STORE(nxt);
    }
    // start global loads for kk+2
    if (kk + 2 < NK) {
#if !HAVE_ASYNC_LDS
      STAGE_A_LOAD(kk + 2);
#endif
      STAGE_B_LOAD(kk + 2);
    }

    // ---- fragments per ISA VGPR layouts ----
    const _Float16* AsB = As[cur];
    const _Float16* BtB = Bt[cur];
    const _Float16* a0p = &AsB[(waveM * 32 + col) * LDA];
    const _Float16* a1p = a0p + 16 * LDA;
    const int ab = hsel ? 8 : 0;
    v8h a0lo = *(const v8h*)(a0p + ab);
    v8h a0hi = *(const v8h*)(a0p + ab + 16);
    v8h a1lo = *(const v8h*)(a1p + ab);
    v8h a1hi = *(const v8h*)(a1p + ab + 16);
    v16h fa0 = __builtin_shufflevector(a0lo, a0hi, 0,1,2,3,4,5,6,7,8,9,10,11,12,13,14,15);
    v16h fa1 = __builtin_shufflevector(a1lo, a1hi, 0,1,2,3,4,5,6,7,8,9,10,11,12,13,14,15);

    const _Float16* b0p = &BtB[(waveN * 32 + col) * LDB];
    const _Float16* b1p = b0p + 16 * LDB;
    const int bb = hsel ? 8 : 0;
    v8h b0lo = *(const v8h*)(b0p + bb);
    v8h b0hi = *(const v8h*)(b0p + bb + 16);
    v8h b1lo = *(const v8h*)(b1p + bb);
    v8h b1hi = *(const v8h*)(b1p + bb + 16);
    v16h fb0 = __builtin_shufflevector(b0lo, b0hi, 0,1,2,3,4,5,6,7,8,9,10,11,12,13,14,15);
    v16h fb1 = __builtin_shufflevector(b1lo, b1hi, 0,1,2,3,4,5,6,7,8,9,10,11,12,13,14,15);

    // ---- 4 x v_wmma_f32_16x16x32_f16 (full EXEC, uniform control flow) ----
    acc00 = __builtin_amdgcn_wmma_f32_16x16x32_f16(false, fa0, false, fb0, (short)0, acc00, false, false);
    acc01 = __builtin_amdgcn_wmma_f32_16x16x32_f16(false, fa0, false, fb1, (short)0, acc01, false, false);
    acc10 = __builtin_amdgcn_wmma_f32_16x16x32_f16(false, fa1, false, fb0, (short)0, acc10, false, false);
    acc11 = __builtin_amdgcn_wmma_f32_16x16x32_f16(false, fa1, false, fb1, (short)0, acc11, false, false);

    WAIT_ASYNC();
    __syncthreads();
  }

  // ---- epilogue: BN + uint8 fake-quant ReLU, coalesced stores ----
  const float step = act_scale[0] * (1.0f / 255.0f);
  const float inv_step = 1.0f / step;

#pragma unroll
  for (int tm = 0; tm < 2; ++tm) {
#pragma unroll
    for (int tn = 0; tn < 2; ++tn) {
      const v8f acc = (tm == 0) ? ((tn == 0) ? acc00 : acc01)
                                : ((tn == 0) ? acc10 : acc11);
      const int coB = coBlk + waveM * 32 + tm * 16 + hsel * 8;
      const int pos = waveN * 32 + tn * 16 + col;
#pragma unroll
      for (int r = 0; r < 8; ++r) {
        const int co = coB + r;
        float y = acc[r] * bnscale[co] + bnbias[co];
        float q = fminf(fmaxf(rintf(y * inv_step), 0.0f), 255.0f) * step;
        out[(((size_t)n * C2 + co) * HDIM + h) * WDIM + pos] = q;
      }
    }
  }
}

// ---------------------------------------------------------------------------
extern "C" void kernel_launch(void* const* d_in, const int* in_sizes, int n_in,
                              void* d_out, int out_size, void* d_ws, size_t ws_size,
                              hipStream_t stream) {
  (void)in_sizes; (void)n_in; (void)out_size; (void)ws_size;
  const float* x     = (const float*)d_in[0];
  const float* w     = (const float*)d_in[1];
  const float* gamma = (const float*)d_in[2];
  const float* beta  = (const float*)d_in[3];
  const float* mean  = (const float*)d_in[4];
  const float* var   = (const float*)d_in[5];
  const float* ascal = (const float*)d_in[6];
  float* out = (float*)d_out;

  _Float16* wpack = (_Float16*)d_ws;                        // 512*2304 f16
  float* bnscale  = (float*)((char*)d_ws + (size_t)C2 * KTOT * sizeof(_Float16));
  float* bnbias   = bnscale + C2;

  qconv_prep_kernel<<<dim3(C2), dim3(256), 0, stream>>>(
      w, gamma, beta, mean, var, wpack, bnscale, bnbias);

  dim3 grid(NB * HDIM, C2 / BLK_M);   // 1024 (n,h) rows x 4 co-blocks
  qconv_main_kernel<<<grid, dim3(256), 0, stream>>>(
      x, wpack, bnscale, bnbias, ascal, out);
}
